// Q_Mlp_8504035246456
// MI455X (gfx1250) — 8-bit matrix operand order repaired and checked by reading, compiled, not run on hardware
//
#include <hip/hip_runtime.h>
#include <hip/hip_bf16.h>
#include <stdint.h>

typedef __attribute__((ext_vector_type(8))) int v8i;

static constexpr size_t OFF_MAX = 0;
static constexpr size_t OFF_W1Q = 256;
static constexpr size_t OFF_W2Q = OFF_W1Q + 3072u * 768u;
static constexpr size_t OFF_BQ1 = OFF_W2Q + 768u * 3072u;
static constexpr size_t OFF_BQ2 = OFF_BQ1 + 3072u * 4u;
static constexpr size_t OFF_X8  = ((OFF_BQ2 + 768u * 4u + 255u) / 256u) * 256u;
static constexpr size_t OFF_H1  = ((OFF_X8 + 12544u * 768u + 255u) / 256u) * 256u;

static constexpr int Mtot = 12544;
static constexpr int N1 = 3072, K1 = 768;
static constexpr int N2 = 768,  K2 = 3072;


__global__ void zero_scales_kernel(unsigned* p) {
    p[0] = 0u; p[1] = 0u;
}

__global__ __launch_bounds__(256)
void maxabs_kernel(const float* __restrict__ W, int n, unsigned* __restrict__ out) {
    float m = 0.0f;
    for (int i = blockIdx.x * blockDim.x + threadIdx.x; i < n;
         i += gridDim.x * blockDim.x)
        m = fmaxf(m, fabsf(W[i]));
    for (int off = 16; off > 0; off >>= 1)
        m = fmaxf(m, __shfl_xor(m, off, 32));
    __shared__ float smax[8];
    const int lane = threadIdx.x & 31, wave = threadIdx.x >> 5;
    if (lane == 0) smax[wave] = m;
    __syncthreads();
    if (threadIdx.x == 0) {
        float mm = smax[0];
        for (int w = 1; w < 8; ++w) mm = fmaxf(mm, smax[w]);
        atomicMax(out, __float_as_uint(mm));
    }
}

__global__ __launch_bounds__(256)
void quantize_w_kernel(const float* __restrict__ W, int8_t* __restrict__ Wq, int n,
                       const unsigned* __restrict__ maxbits) {
    const float s = __uint_as_float(*maxbits) * (1.0f / 127.0f);
    for (int i = blockIdx.x * blockDim.x + threadIdx.x; i < n;
         i += gridDim.x * blockDim.x) {
        float q = rintf(W[i] / s);
        q = fminf(fmaxf(q, -127.0f), 127.0f);
        Wq[i] = (int8_t)(int)q;
    }
}

__global__ __launch_bounds__(256)
void quantize_x_kernel(const float* __restrict__ x, int8_t* __restrict__ x8, int n) {
    for (int i = blockIdx.x * blockDim.x + threadIdx.x; i < n;
         i += gridDim.x * blockDim.x)
        x8[i] = (int8_t)(int)rintf(x[i]);
}

__global__ __launch_bounds__(256)
void bias_quant_kernel(const float* __restrict__ b, int* __restrict__ bq, int n,
                       const unsigned* __restrict__ maxbits,
                       const float* __restrict__ a_prev) {
    const float bscale = (__uint_as_float(*maxbits) * (1.0f / 127.0f)) * (*a_prev);
    for (int i = blockIdx.x * blockDim.x + threadIdx.x; i < n;
         i += gridDim.x * blockDim.x)
        bq[i] = (int)rintf(b[i] / bscale);
}

__global__ void write_scale_kernel(float* dst, const float* a2) { *dst = *a2; }

__device__ __forceinline__ v8i load_a_frag(const int8_t* p) {
    int2 t0 = *(const int2*)(p + 0);
    int2 t1 = *(const int2*)(p + 16);
    int2 t2 = *(const int2*)(p + 32);
    int2 t3 = *(const int2*)(p + 48);
    v8i r;
    r[0] = t0.x; r[1] = t0.y; r[2] = t1.x; r[3] = t1.y;
    r[4] = t2.x; r[5] = t2.y; r[6] = t3.x; r[7] = t3.y;
    return r;
}

__device__ __forceinline__ v8i load_b_frag(const int8_t* p) {
    int4 t0 = *(const int4*)(p + 0);
    int4 t1 = *(const int4*)(p + 32);
    v8i r;
    r[0] = t0.x; r[1] = t0.y; r[2] = t0.z; r[3] = t0.w;
    r[4] = t1.x; r[5] = t1.y; r[6] = t1.z; r[7] = t1.w;
    return r;
}

__global__ __launch_bounds__(256)
void wmma_gemm_i8_kernel(const int8_t* __restrict__ A, const int8_t* __restrict__ Bm,
                         const int* __restrict__ bq, int N, int K,
                         const unsigned* __restrict__ maxbits,
                         const float* __restrict__ s_in, const float* __restrict__ s_out,
                         int8_t* __restrict__ out8, float* __restrict__ outf) {
    const int lane = threadIdx.x & 31;
    const int wave = threadIdx.x >> 5;
    const int wm = wave & 3;
    const int wn = wave >> 2;
    const int m0 = blockIdx.y * 128 + wm * 32;
    const int n0 = blockIdx.x * 128 + wn * 64;

    const int row = lane & 15;
    const int hi = (lane >= 16) ? 1 : 0;

    const int8_t* a_ptr0 = A + (size_t)(m0 + row) * K + (hi ? 8 : 0);
    const int8_t* a_ptr1 = a_ptr0 + (size_t)16 * K;
    const int8_t* b_base = Bm + (size_t)(n0 + row) * K + (hi ? 8 : 0);

    v8i c[2][4];
#pragma unroll
    for (int i = 0; i < 2; ++i)
#pragma unroll
        for (int j = 0; j < 4; ++j)
            c[i][j] = (v8i){};

    for (int k0 = 0; k0 < K; k0 += 64) {
        v8i a0 = load_a_frag(a_ptr0 + k0);
        v8i a1 = load_a_frag(a_ptr1 + k0);
        v8i b0 = load_a_frag(b_base + k0);
        v8i b1 = load_a_frag(b_base + (size_t)16 * K + k0);
        v8i b2 = load_a_frag(b_base + (size_t)32 * K + k0);
        v8i b3 = load_a_frag(b_base + (size_t)48 * K + k0);
        c[0][0] = __builtin_amdgcn_wmma_i32_16x16x64_iu8(true, a0, true, b0, c[0][0], false, false);
        c[0][1] = __builtin_amdgcn_wmma_i32_16x16x64_iu8(true, a0, true, b1, c[0][1], false, false);
        c[0][2] = __builtin_amdgcn_wmma_i32_16x16x64_iu8(true, a0, true, b2, c[0][2], false, false);
        c[0][3] = __builtin_amdgcn_wmma_i32_16x16x64_iu8(true, a0, true, b3, c[0][3], false, false);
        c[1][0] = __builtin_amdgcn_wmma_i32_16x16x64_iu8(true, a1, true, b0, c[1][0], false, false);
        c[1][1] = __builtin_amdgcn_wmma_i32_16x16x64_iu8(true, a1, true, b1, c[1][1], false, false);
        c[1][2] = __builtin_amdgcn_wmma_i32_16x16x64_iu8(true, a1, true, b2, c[1][2], false, false);
        c[1][3] = __builtin_amdgcn_wmma_i32_16x16x64_iu8(true, a1, true, b3, c[1][3], false, false);
    }

    const float mult = ((__uint_as_float(*maxbits) * (1.0f / 127.0f)) * (*s_in)) / (*s_out);

    const int col0 = n0 + row;
    int bias[4];
#pragma unroll
    for (int j = 0; j < 4; ++j) bias[j] = bq[col0 + 16 * j];
    const int rbase = m0 + (hi ? 8 : 0);

    if (out8) {
#pragma unroll
        for (int i = 0; i < 2; ++i) {
            const int rb = rbase + 16 * i;
#pragma unroll
            for (int j = 0; j < 4; ++j) {
                const int col = col0 + 16 * j;
#pragma unroll
                for (int r = 0; r < 8; ++r) {
                    float v = rintf((float)(c[i][j][r] + bias[j]) * mult);
                    v = fminf(fmaxf(v, 0.0f), 127.0f);
                    out8[(size_t)(rb + r) * N + col] = (int8_t)(int)v;
                }
            }
        }
    } else {
#pragma unroll
        for (int i = 0; i < 2; ++i) {
            const int rb = rbase + 16 * i;
#pragma unroll
            for (int j = 0; j < 4; ++j) {
                const int col = col0 + 16 * j;
#pragma unroll
                for (int r = 0; r < 8; ++r) {
                    float v = rintf((float)(c[i][j][r] + bias[j]) * mult);
                    v = fminf(fmaxf(v, -128.0f), 127.0f);
                    outf[(size_t)(rb + r) * N + col] = v;
                }
            }
        }
    }
}

extern "C" void kernel_launch(void* const* d_in, const int* in_sizes, int n_in,
                              void* d_out, int out_size, void* d_ws, size_t ws_size,
                              hipStream_t stream) {
    const float* x   = (const float*)d_in[0];
    const float* a_s = (const float*)d_in[1];
    const float* W1  = (const float*)d_in[2];
    const float* b1  = (const float*)d_in[3];
    const float* W2  = (const float*)d_in[4];
    const float* b2  = (const float*)d_in[5];
    const float* a1  = (const float*)d_in[6];
    const float* a2  = (const float*)d_in[7];
    float* out = (float*)d_out;

    char* ws = (char*)d_ws;
    unsigned* maxb = (unsigned*)(ws + OFF_MAX);
    int8_t* W1q = (int8_t*)(ws + OFF_W1Q);
    int8_t* W2q = (int8_t*)(ws + OFF_W2Q);
    int*    bq1 = (int*)(ws + OFF_BQ1);
    int*    bq2 = (int*)(ws + OFF_BQ2);
    int8_t* x8  = (int8_t*)(ws + OFF_X8);
    int8_t* h1  = (int8_t*)(ws + OFF_H1);

    const int nW1 = N1 * K1;
    const int nW2 = N2 * K2;
    const int nX  = Mtot * K1;

    zero_scales_kernel<<<1, 1, 0, stream>>>(maxb);

    maxabs_kernel<<<dim3(1024), dim3(256), 0, stream>>>(W1, nW1, maxb + 0);
    maxabs_kernel<<<dim3(1024), dim3(256), 0, stream>>>(W2, nW2, maxb + 1);

    quantize_w_kernel<<<dim3(2048), dim3(256), 0, stream>>>(W1, W1q, nW1, maxb + 0);
    quantize_w_kernel<<<dim3(2048), dim3(256), 0, stream>>>(W2, W2q, nW2, maxb + 1);
    quantize_x_kernel<<<dim3(4096), dim3(256), 0, stream>>>(x, x8, nX);

    bias_quant_kernel<<<dim3(12), dim3(256), 0, stream>>>(b1, bq1, N1, maxb + 0, a_s);
    bias_quant_kernel<<<dim3(3),  dim3(256), 0, stream>>>(b2, bq2, N2, maxb + 1, a1);

    {
        dim3 grid(N1 / 128, Mtot / 128);
        wmma_gemm_i8_kernel<<<grid, dim3(256), 0, stream>>>(
            x8, W1q, bq1, N1, K1, maxb + 0, a_s, a1, h1, nullptr);
    }
    {
        dim3 grid(N2 / 128, Mtot / 128);
        wmma_gemm_i8_kernel<<<grid, dim3(256), 0, stream>>>(
            h1, W2q, bq2, N2, K2, maxb + 1, a1, a2, nullptr, out);
    }
    write_scale_kernel<<<1, 1, 0, stream>>>(out + (size_t)Mtot * N2, a2);
}
